// TeamCommAgent_53077205844655
// MI455X (gfx1250) — matrix operand order repaired and checked by reading, not compiled, not run on hardware
//
#include <hip/hip_runtime.h>
#include <math.h>

// Problem constants (from reference): B=4096, A=32, OBS=128, HID=256, MSG=64, NH=4, NACT=16
#define NBATCH 4096
#define NAGENT 32
#define NOBS   128
#define NHID   256
#define NMSG   64
#define NACT   16
#define MTOK   (NBATCH * NAGENT)   // 131072 tokens

typedef __attribute__((ext_vector_type(16))) __bf16 v16bf;
typedef __attribute__((ext_vector_type(8)))  __bf16 v8bf;
typedef __attribute__((ext_vector_type(8)))  float  v8f;

// ---------------------------------------------------------------------------
// fp32 -> bf16 bulk convert (one-time pass for obs + weights; emits v_cvt_pk)
// ---------------------------------------------------------------------------
__global__ __launch_bounds__(256) void cvt_f32_bf16(const float* __restrict__ in,
                                                    __bf16* __restrict__ out, int n)
{
    const int i = blockIdx.x * 256 + threadIdx.x;
    if (i < n) out[i] = (__bf16)in[i];
}

// ---------------------------------------------------------------------------
// One 32-deep k-step of the 2x2-tile WMMA GEMM.  All operands bf16 in memory,
// loaded with aligned vector loads; zero per-element conversion work.
// ---------------------------------------------------------------------------
__device__ __forceinline__ void k32_step(
    const __bf16*& p0, const __bf16*& p1,
    const __bf16*& wp0, const __bf16*& wp1,
    v8f& acc00, v8f& acc01, v8f& acc10, v8f& acc11)
{
    const v8bf a0lo = *(const v8bf*)p0;            // k offsets 0..7   (per lane-half)
    const v8bf a0hi = *(const v8bf*)(p0 + 16);     // k offsets 16..23
    const v8bf a1lo = *(const v8bf*)p1;
    const v8bf a1hi = *(const v8bf*)(p1 + 16);
    const v16bf a0 = __builtin_shufflevector(a0lo, a0hi, 0,1,2,3,4,5,6,7,8,9,10,11,12,13,14,15);
    const v16bf a1 = __builtin_shufflevector(a1lo, a1hi, 0,1,2,3,4,5,6,7,8,9,10,11,12,13,14,15);
    const v16bf b0 = __builtin_shufflevector(*(const v8bf*)wp0, *(const v8bf*)(wp0 + 16), 0,1,2,3,4,5,6,7,8,9,10,11,12,13,14,15);  // k offsets 0..7, 16..23 (per lane-half)
    const v16bf b1 = __builtin_shufflevector(*(const v8bf*)wp1, *(const v8bf*)(wp1 + 16), 0,1,2,3,4,5,6,7,8,9,10,11,12,13,14,15);

    acc00 = __builtin_amdgcn_wmma_f32_16x16x32_bf16(false, a0, false, b0, (short)0, acc00, false, false);
    acc01 = __builtin_amdgcn_wmma_f32_16x16x32_bf16(false, a0, false, b1, (short)0, acc01, false, false);
    acc10 = __builtin_amdgcn_wmma_f32_16x16x32_bf16(false, a1, false, b0, (short)0, acc10, false, false);
    acc11 = __builtin_amdgcn_wmma_f32_16x16x32_bf16(false, a1, false, b1, (short)0, acc11, false, false);

    p0 += 32; p1 += 32; wp0 += 32; wp1 += 32;
}

// ---------------------------------------------------------------------------
// WMMA GEMM, 2x2 tiles of 16x16 per wave (32x32 output per wave), bf16 in/out,
// f32 accumulate:   out[m,n] = act(m,:) . W[n,:] + bias[n]   (+ optional tanh)
//
// Activation "act" is a virtual concat of three bf16 sources, split into three
// straight-line k-segment loops (boundaries c1,c2 multiples of 32):
//   k <  c1          -> A1[m*ld1 + k]
//   c1 <= k < c2     -> A2[m*ld2 + (k-c1)]
//   k >= c2          -> A3[(m>>5)*ld3 + (k-c2)]     (per-batch broadcast, A=32)
// W is bf16 row-major [N,K]. K%32==0, N%32==0, M%32==0. Pass c1=c2=K for a
// plain single-source GEMM (segments 2/3 are zero-trip).
//
// Fragment layouts per CDNA5 ISA (16-bit, wave32):
//   A 16x32: lane L holds row L&15; element i -> k = (i&7) + 16*(i>>3) + 8*(L>>4)
//   B 32x16: lane L holds col L&15; element i -> k = (i&7) + 16*(i>>3) + 8*(L>>4)
//   C 16x16: lane L, vgpr r -> row r + 8*(L>>4), col L&15
// ---------------------------------------------------------------------------
template <int DO_TANH>
__global__ __launch_bounds__(256) void gemm_bf16_wmma(
    const __bf16* __restrict__ A1, int ld1, int c1,
    const __bf16* __restrict__ A2, int ld2, int c2,
    const __bf16* __restrict__ A3, int ld3,
    const __bf16* __restrict__ W,  const float* __restrict__ bias,
    __bf16* __restrict__ out,
    int Mt2, int Nt2, int K, int ldo)
{
    const int wave = threadIdx.x >> 5;
    const int lane = threadIdx.x & 31;
    const long long t = (long long)blockIdx.x * 8 + wave;
    if (t >= (long long)Mt2 * Nt2) return;        // wave-uniform exit: EXEC all-1s for WMMA
    const int mt = (int)(t % Mt2);                // 32-row tile index
    const int nt = (int)(t / Mt2);                // 32-col tile index

    const int lh  = lane >> 4;                    // lane half
    const int l15 = lane & 15;
    const int m0  = mt * 32 + l15;                // A rows owned by this lane
    const int m1  = m0 + 16;
    const int n0  = nt * 32 + l15;                // B/C/D cols owned by this lane
    const int n1  = n0 + 16;

    v8f acc00 = {}, acc01 = {}, acc10 = {}, acc11 = {};
    const __bf16* wp0 = W + (long long)n0 * K + 8 * lh;
    const __bf16* wp1 = W + (long long)n1 * K + 8 * lh;

    // Segment 1: A1 rows
    {
        const __bf16* p0 = A1 + (long long)m0 * ld1 + 8 * lh;
        const __bf16* p1 = A1 + (long long)m1 * ld1 + 8 * lh;
        const int e = (c1 < K) ? c1 : K;
        for (int kk = 0; kk < e; kk += 32)
            k32_step(p0, p1, wp0, wp1, acc00, acc01, acc10, acc11);
    }
    // Segment 2: A2 rows
    if (c2 > c1) {
        const __bf16* p0 = A2 + (long long)m0 * ld2 + 8 * lh;
        const __bf16* p1 = A2 + (long long)m1 * ld2 + 8 * lh;
        for (int kk = c1; kk < c2; kk += 32)
            k32_step(p0, p1, wp0, wp1, acc00, acc01, acc10, acc11);
    }
    // Segment 3: A3 rows (per-batch broadcast; 32-aligned row blocks share a batch)
    if (K > c2) {
        const __bf16* p0 = A3 + (long long)(m0 >> 5) * ld3 + 8 * lh;
        const __bf16* p1 = A3 + (long long)(m1 >> 5) * ld3 + 8 * lh;
        for (int kk = c2; kk < K; kk += 32)
            k32_step(p0, p1, wp0, wp1, acc00, acc01, acc10, acc11);
    }

    const float bv0 = bias[n0];
    const float bv1 = bias[n1];
#pragma unroll
    for (int r = 0; r < 8; ++r) {
        const int r0 = mt * 32 + r + (lh << 3);        // rows of tile 0
        const int r1 = r0 + 16;                        // rows of tile 1
        float y00 = acc00[r] + bv0, y01 = acc01[r] + bv1;
        float y10 = acc10[r] + bv0, y11 = acc11[r] + bv1;
        if (DO_TANH) { y00 = tanhf(y00); y01 = tanhf(y01); y10 = tanhf(y10); y11 = tanhf(y11); }
        out[(long long)r0 * ldo + n0] = (__bf16)y00;
        out[(long long)r0 * ldo + n1] = (__bf16)y01;
        out[(long long)r1 * ldo + n0] = (__bf16)y10;
        out[(long long)r1 * ldo + n1] = (__bf16)y11;
    }
}

// ---------------------------------------------------------------------------
// Multi-head self-attention core over A=32 agents, MSG=64, 4 heads of dh=16.
// qkv: bf16 [B*A, 192] (q|k|v).  o: bf16 [B*A, 64] = concat-head output (pre-Wo).
// One block per batch; thread = (head, query).  Math in f32.
// ---------------------------------------------------------------------------
__global__ __launch_bounds__(128) void mha_core(const __bf16* __restrict__ qkv,
                                                __bf16* __restrict__ o)
{
    __shared__ float ks[NAGENT][NMSG];
    __shared__ float vs[NAGENT][NMSG];
    const int b = blockIdx.x;
    const __bf16* base = qkv + (long long)b * NAGENT * 192;

    for (int i = threadIdx.x; i < NAGENT * NMSG; i += 128) {
        const int a = i >> 6, c = i & 63;
        ks[a][c] = (float)base[a * 192 + 64 + c];
        vs[a][c] = (float)base[a * 192 + 128 + c];
    }
    __syncthreads();

    const int h = threadIdx.x >> 5;
    const int q = threadIdx.x & 31;

    float qv[16];
#pragma unroll
    for (int i = 0; i < 16; ++i) qv[i] = (float)base[q * 192 + h * 16 + i];

    float s[NAGENT];
    float mx = -1e30f;
    for (int a = 0; a < NAGENT; ++a) {
        float d = 0.f;
#pragma unroll
        for (int i = 0; i < 16; ++i) d += qv[i] * ks[a][h * 16 + i];
        d *= 0.25f;                                // 1/sqrt(dh=16)
        s[a] = d;
        mx = fmaxf(mx, d);
    }
    float sum = 0.f;
    for (int a = 0; a < NAGENT; ++a) { s[a] = expf(s[a] - mx); sum += s[a]; }
    const float inv = 1.f / sum;

    float ov[16] = {};
    for (int a = 0; a < NAGENT; ++a) {
        const float p = s[a];
#pragma unroll
        for (int i = 0; i < 16; ++i) ov[i] += p * vs[a][h * 16 + i];
    }
    __bf16* orow = o + (long long)(b * NAGENT + q) * NMSG + h * 16;
#pragma unroll
    for (int i = 0; i < 16; ++i) orow[i] = (__bf16)(ov[i] * inv);
}

// mu[i] += softplus(stdp[i] - 5) * eps[i]   (in place -> becomes intra/inter obs)
__global__ __launch_bounds__(256) void reparam(__bf16* __restrict__ mu,
                                               const __bf16* __restrict__ stdp,
                                               const float* __restrict__ eps, int n)
{
    const int i = blockIdx.x * 256 + threadIdx.x;
    if (i >= n) return;
    const float x = (float)stdp[i] - 5.0f;
    const float sp = (x > 20.f) ? x : log1pf(expf(x));
    mu[i] = (__bf16)((float)mu[i] + sp * eps[i]);
}

// Per-batch softmax pooling over agents: pooled[b,:] = sum_a softmax_a(x.attw+attb) * x[a,:]
__global__ __launch_bounds__(64) void pool(const __bf16* __restrict__ inter_obs,
                                           const float* __restrict__ att_w,
                                           const float* __restrict__ att_b,
                                           __bf16* __restrict__ pooled)
{
    __shared__ float row[NAGENT][NMSG];
    __shared__ float p[NAGENT];
    const int b = blockIdx.x;
    const __bf16* base = inter_obs + (long long)b * NAGENT * NMSG;

    for (int i = threadIdx.x; i < NAGENT * NMSG; i += 64) row[i >> 6][i & 63] = (float)base[i];
    __syncthreads();

    if (threadIdx.x < NAGENT) {
        const int a = threadIdx.x;
        float acc = att_b[0];
        for (int c = 0; c < NMSG; ++c) acc += row[a][c] * att_w[c];
        p[a] = acc;
    }
    __syncthreads();
    if (threadIdx.x == 0) {
        float mx = -1e30f;
        for (int a = 0; a < NAGENT; ++a) mx = fmaxf(mx, p[a]);
        float s = 0.f;
        for (int a = 0; a < NAGENT; ++a) { p[a] = expf(p[a] - mx); s += p[a]; }
        const float inv = 1.f / s;
        for (int a = 0; a < NAGENT; ++a) p[a] *= inv;
    }
    __syncthreads();
    const int c = threadIdx.x;                     // 0..63
    float acc = 0.f;
    for (int a = 0; a < NAGENT; ++a) acc += p[a] * row[a][c];
    pooled[(long long)b * NMSG + c] = (__bf16)acc;
}

// Final heads: a = log_softmax(h2 @ Wa^T + ba), v = h2 @ Wv^T + bv
// d_out layout: [M*16] f32 action log-probs, then [M] f32 values.
__global__ __launch_bounds__(256) void head(const __bf16* __restrict__ h2,
                                            const float* __restrict__ Wa,
                                            const float* __restrict__ ba,
                                            const float* __restrict__ Wv,
                                            const float* __restrict__ bv,
                                            float* __restrict__ out, int M)
{
    const int m = blockIdx.x * 256 + threadIdx.x;
    if (m >= M) return;
    const __bf16* hr = h2 + (long long)m * NHID;

    float logits[NACT];
    float mx = -1e30f;
    for (int n = 0; n < NACT; ++n) {
        const float* wr = Wa + n * NHID;
        float acc = ba[n];
        for (int k = 0; k < NHID; ++k) acc += (float)hr[k] * wr[k];
        logits[n] = acc;
        mx = fmaxf(mx, acc);
    }
    float vacc = bv[0];
    for (int k = 0; k < NHID; ++k) vacc += (float)hr[k] * Wv[k];

    float s = 0.f;
    for (int n = 0; n < NACT; ++n) s += expf(logits[n] - mx);
    const float lse = mx + logf(s);
    for (int n = 0; n < NACT; ++n) out[(long long)m * NACT + n] = logits[n] - lse;
    out[(long long)M * NACT + m] = vacc;
}

// ---------------------------------------------------------------------------
extern "C" void kernel_launch(void* const* d_in, const int* in_sizes, int n_in,
                              void* d_out, int out_size, void* d_ws, size_t ws_size,
                              hipStream_t stream) {
    (void)in_sizes; (void)n_in; (void)out_size; (void)ws_size;

    const float* obs       = (const float*)d_in[0];
    const float* eps_intra = (const float*)d_in[1];
    const float* eps_inter = (const float*)d_in[2];
    const float* W_local   = (const float*)d_in[3];
    const float* b_local   = (const float*)d_in[4];
    const float* W_inter   = (const float*)d_in[5];
    const float* b_inter   = (const float*)d_in[6];
    const float* W_intra   = (const float*)d_in[7];
    const float* b_intra   = (const float*)d_in[8];
    const float* Wqkv      = (const float*)d_in[9];   // [4,192,64]
    const float* bqkv      = (const float*)d_in[10];  // [4,192]
    const float* Wo        = (const float*)d_in[11];  // [4,64,64]
    const float* bo        = (const float*)d_in[12];  // [4,64]
    const float* att_w     = (const float*)d_in[13];
    const float* att_b     = (const float*)d_in[14];
    const float* W1        = (const float*)d_in[15];  // [256,384]
    const float* b1        = (const float*)d_in[16];
    const float* W2        = (const float*)d_in[17];  // [256,256]
    const float* b2        = (const float*)d_in[18];
    const float* Wa        = (const float*)d_in[19];  // [16,256]
    const float* ba        = (const float*)d_in[20];
    const float* Wv        = (const float*)d_in[21];  // [1,256]
    const float* bv        = (const float*)d_in[22];

    float* out = (float*)d_out;
    __bf16* bw = (__bf16*)d_ws;

    const long long M = MTOK;
    // Workspace carve-up (bf16 elements), with aliasing:
    __bf16* obs_bf  = bw;                        // [M,128]
    __bf16* local   = obs_bf  + M * NOBS;        // [M,256]  (reused as h2 later)
    __bf16* intra_e = local   + M * NHID;        // [M,64]
    __bf16* inter_e = intra_e + M * NMSG;        // [M,64]
    __bf16* att0    = inter_e + M * NMSG;        // [M,64] -> intra_obs (in place)
    __bf16* att1    = att0    + M * NMSG;        // [M,64]
    __bf16* att2    = att1    + M * NMSG;        // [M,64] -> inter_obs (in place)
    __bf16* att3    = att2    + M * NMSG;        // [M,64]
    __bf16* qkvbuf  = att3    + M * NMSG;        // [M,192] (reused as h1 [M,256])
    __bf16* attno   = qkvbuf  + M * 192;         // [M,64]
    __bf16* h1      = qkvbuf;                    // alias: qkv region free after attn stages
    __bf16* h2      = local;                     // alias: local free after W1 GEMM
    __bf16* pooled  = qkvbuf  + M * NHID;        // [B,64]
    // bf16 weight copies (tiny, converted once per launch):
    __bf16* wb_local = pooled   + (long long)NBATCH * NMSG;   // 256*128
    __bf16* wb_inter = wb_local + NHID * NOBS;                // 64*128
    __bf16* wb_intra = wb_inter + NMSG * NOBS;                // 64*128
    __bf16* wb_qkv   = wb_intra + NMSG * NOBS;                // 4*192*64
    __bf16* wb_wo    = wb_qkv   + 4 * 192 * NMSG;             // 4*64*64
    __bf16* wb_w1    = wb_wo    + 4 * NMSG * NMSG;            // 256*384
    __bf16* wb_w2    = wb_w1    + NHID * (NHID + 2 * NMSG);   // 256*256

    auto cvt = [&](const float* src, __bf16* dst, int n) {
        cvt_f32_bf16<<<(unsigned)((n + 255) / 256), 256, 0, stream>>>(src, dst, n);
    };
    // 0) One-time precision conversion: obs + all WMMA-consumed weights -> bf16
    cvt(obs,     obs_bf,   (int)(M * NOBS));
    cvt(W_local, wb_local, NHID * NOBS);
    cvt(W_inter, wb_inter, NMSG * NOBS);
    cvt(W_intra, wb_intra, NMSG * NOBS);
    cvt(Wqkv,    wb_qkv,   4 * 192 * NMSG);
    cvt(Wo,      wb_wo,    4 * NMSG * NMSG);
    cvt(W1,      wb_w1,    NHID * (NHID + 2 * NMSG));
    cvt(W2,      wb_w2,    NHID * NHID);

    const int Mt2 = (int)(M / 32);              // 4096 32-row tiles
    auto gblocks = [](long long tiles) { return (unsigned)((tiles + 7) / 8); };

    // 1) Embeddings: tanh(obs @ W^T + b)
    gemm_bf16_wmma<1><<<gblocks((long long)Mt2 * (NHID / 32)), 256, 0, stream>>>(
        obs_bf, NOBS, NOBS, obs_bf, 0, NOBS, obs_bf, 0, wb_local, b_local,
        local, Mt2, NHID / 32, NOBS, NHID);
    gemm_bf16_wmma<1><<<gblocks((long long)Mt2 * (NMSG / 32)), 256, 0, stream>>>(
        obs_bf, NOBS, NOBS, obs_bf, 0, NOBS, obs_bf, 0, wb_inter, b_inter,
        inter_e, Mt2, NMSG / 32, NOBS, NMSG);
    gemm_bf16_wmma<1><<<gblocks((long long)Mt2 * (NMSG / 32)), 256, 0, stream>>>(
        obs_bf, NOBS, NOBS, obs_bf, 0, NOBS, obs_bf, 0, wb_intra, b_intra,
        intra_e, Mt2, NMSG / 32, NOBS, NMSG);

    // 2) Four attention nets (sequential; qkv/attno buffers reused)
    __bf16* attout[4] = { att0, att1, att2, att3 };
    for (int g = 0; g < 4; ++g) {
        const __bf16* e = (g < 2) ? intra_e : inter_e;
        // qkv = e @ Wqkv[g]^T + bqkv[g]   [M,192]
        gemm_bf16_wmma<0><<<gblocks((long long)Mt2 * (192 / 32)), 256, 0, stream>>>(
            e, NMSG, NMSG, e, 0, NMSG, e, 0,
            wb_qkv + (long long)g * 192 * NMSG, bqkv + g * 192,
            qkvbuf, Mt2, 192 / 32, NMSG, 192);
        // softmax attention core
        mha_core<<<NBATCH, 128, 0, stream>>>(qkvbuf, attno);
        // out proj: att[g] = attno @ Wo[g]^T + bo[g]
        gemm_bf16_wmma<0><<<gblocks((long long)Mt2 * (NMSG / 32)), 256, 0, stream>>>(
            attno, NMSG, NMSG, attno, 0, NMSG, attno, 0,
            wb_wo + (long long)g * NMSG * NMSG, bo + g * NMSG,
            attout[g], Mt2, NMSG / 32, NMSG, NMSG);
    }

    // 3) Reparameterize (in place): att0 -> intra_obs, att2 -> inter_obs
    {
        const int n = (int)(M * NMSG);
        reparam<<<(n + 255) / 256, 256, 0, stream>>>(att0, att1, eps_intra, n);
        reparam<<<(n + 255) / 256, 256, 0, stream>>>(att2, att3, eps_inter, n);
    }

    // 4) Softmax pooling over agents
    pool<<<NBATCH, 64, 0, stream>>>(att2, att_w, att_b, pooled);

    // 5) h1 = tanh([local | intra_obs | pooled_b] @ W1^T + b1)  (virtual concat, K=384;
    //    boundaries 256/320 are multiples of 32 -> straight-line segment loops)
    gemm_bf16_wmma<1><<<gblocks((long long)Mt2 * (NHID / 32)), 256, 0, stream>>>(
        local, NHID, NHID, att0, NMSG, NHID + NMSG, pooled, NMSG,
        wb_w1, b1, h1, Mt2, NHID / 32, NHID + 2 * NMSG, NHID);

    // 6) h2 = tanh(h1 @ W2^T + b2)   (writes over local buffer)
    gemm_bf16_wmma<1><<<gblocks((long long)Mt2 * (NHID / 32)), 256, 0, stream>>>(
        h1, NHID, NHID, h1, 0, NHID, h1, 0,
        wb_w2, b2, h2, Mt2, NHID / 32, NHID, NHID);

    // 7) Action log-softmax + value heads -> d_out (f32)
    head<<<(unsigned)((M + 255) / 256), 256, 0, stream>>>(h2, Wa, ba, Wv, bv, out, (int)M);
}
